// RQCodebook_67010079752343
// MI455X (gfx1250) — matrix operand order repaired and checked by reading, not compiled, not run on hardware
//
#include <hip/hip_runtime.h>
#include <hip/hip_bf16.h>
#include <stdint.h>

typedef __bf16 v16bf __attribute__((ext_vector_type(16)));
typedef float  v8f   __attribute__((ext_vector_type(8)));
typedef unsigned short v16us __attribute__((ext_vector_type(16)));

#define D_DIM 128
#define K_CODES 1024
#define L_LEVELS 4
#define ROWS_PER_BLOCK 64
#define PITCH 132   // 128 + 4 dword pad -> conflict-free row-strided LDS access
#define THREADS 128
#define NTILES (K_CODES / 16)

static __device__ inline v8f wmma_bf16(v16bf a, v16bf b, v8f c) {
  return __builtin_amdgcn_wmma_f32_16x16x32_bf16(false, a, false, b, (short)0, c, false, false);
}

// split (-x) into bf16 hi/lo bit patterns (truncation split; hi+lo ~ 16 mantissa bits)
static __device__ inline void split_neg(float x, unsigned short& hb, unsigned short& lb) {
  float a = -x;
  unsigned int u = __float_as_uint(a);
  hb = (unsigned short)(u >> 16);
  float hf = __uint_as_float(u & 0xffff0000u);
  float lo = a - hf;
  lb = (unsigned short)(__float_as_uint(lo) >> 16);
}

// ---- prep: codebooks f32 [L*K, D] -> bf16 hi/lo planes + 0.5*||e||^2
__global__ void rvq_prep_67010079752343(const float* __restrict__ cb,
                                        unsigned short* __restrict__ bhi,
                                        unsigned short* __restrict__ blo,
                                        float* __restrict__ e2) {
  const int row = blockIdx.x;   // l*K + k
  const int d   = threadIdx.x;  // 0..127
  const float v = cb[(size_t)row * D_DIM + d];
  unsigned int u = __float_as_uint(v);
  unsigned short hb = (unsigned short)(u >> 16);
  float hf = __uint_as_float(u & 0xffff0000u);
  float lo = v - hf;
  unsigned short lb = (unsigned short)(__float_as_uint(lo) >> 16);
  bhi[(size_t)row * D_DIM + d] = hb;
  blo[(size_t)row * D_DIM + d] = lb;
  __shared__ float red[128];
  red[d] = v * v;
  __syncthreads();
  for (int s = 64; s > 0; s >>= 1) {
    if (d < s) red[d] += red[d + s];
    __syncthreads();
  }
  if (d == 0) e2[row] = 0.5f * red[0];
}

// ---- main: all 4 RVQ levels in one kernel; residual resident in LDS
__global__ __launch_bounds__(THREADS) void rvq_main_67010079752343(
    const float* __restrict__ res_in, const float* __restrict__ cb,
    const unsigned short* __restrict__ bhi, const unsigned short* __restrict__ blo,
    const float* __restrict__ e2, float* __restrict__ qout, int* __restrict__ codes) {
  __shared__ float s_res[ROWS_PER_BLOCK * PITCH];

  const int tid    = threadIdx.x;
  const int lane   = tid & 31;
  const int wave   = tid >> 5;     // 0..3, each owns a 16-row M tile
  const int h      = lane >> 4;    // lane half (per ISA A/B/C layouts)
  const int nlane  = lane & 15;
  const int m_base = wave * 16;
  const size_t row0 = (size_t)blockIdx.x * ROWS_PER_BLOCK;

  // cooperative residual -> LDS via CDNA5 async copy (ASYNCcnt-tracked, GVS mode)
  {
    const unsigned long long gbase = (unsigned long long)(res_in + row0 * D_DIM);
#pragma unroll
    for (int i = 0; i < (ROWS_PER_BLOCK * D_DIM / 4) / THREADS; ++i) {
      int linear = i * THREADS + tid;  // float4 index
      int r = linear >> 5;             // 32 float4 per row
      int c4 = linear & 31;
      unsigned lds_addr = (unsigned)(uintptr_t)&s_res[r * PITCH + c4 * 4];
      unsigned goff = (unsigned)((r * D_DIM + c4 * 4) * 4);
      asm volatile("global_load_async_to_lds_b128 %0, %1, %2"
                   :
                   : "v"(lds_addr), "v"(goff), "s"(gbase)
                   : "memory");
    }
    asm volatile("s_wait_asynccnt 0" ::: "memory");
  }
  __syncthreads();

#pragma unroll 1
  for (int l = 0; l < L_LEVELS; ++l) {
    // A tile: negated residual, bf16 hi/lo, contiguous K order matching B:
    // lane<16: elems 0..15 = K kk+0..15 ; lane>=16: K kk+16..31
    const int m = m_base + nlane;
    v16bf ahi[4], alo[4];
#pragma unroll
    for (int kki = 0; kki < 4; ++kki) {
      const int kk = kki * 32;
      unsigned short hb[16], lb[16];
      const float* rp = &s_res[m * PITCH];
#pragma unroll
      for (int j = 0; j < 8; ++j) split_neg(rp[kk + h * 16 + j], hb[j], lb[j]);
#pragma unroll
      for (int j = 0; j < 8; ++j) split_neg(rp[kk + 8 + h * 16 + j], hb[8 + j], lb[8 + j]);
      v16us th, tl;
#pragma unroll
      for (int j = 0; j < 16; ++j) { th[j] = hb[j]; tl[j] = lb[j]; }
      ahi[kki] = __builtin_bit_cast(v16bf, th);
      alo[kki] = __builtin_bit_cast(v16bf, tl);
    }

    // per-lane B pointers (B 32x16 layout: lane = column n, lane half selects K block)
    const unsigned short* ph = bhi + (size_t)l * K_CODES * D_DIM + (size_t)nlane * D_DIM + h * 16;
    const unsigned short* pl = blo + (size_t)l * K_CODES * D_DIM + (size_t)nlane * D_DIM + h * 16;
    const float* e2p = e2 + l * K_CODES + nlane;  // tile nt value at e2p[nt*16]

    // software pipeline: preload tile 0's B fragments + e2 into registers
    v16us sh[4], sl[4];
#pragma unroll
    for (int kki = 0; kki < 4; ++kki) {
      sh[kki] = *(const v16us*)(ph + kki * 32);
      sl[kki] = *(const v16us*)(pl + kki * 32);
    }
    float e2cur = e2p[0];

    float vmin[8];
    int vidx[8];
#pragma unroll
    for (int i = 0; i < 8; ++i) { vmin[i] = 3.402823466e+38f; vidx[i] = 0; }

#pragma unroll 1
    for (int nt = 0; nt < NTILES; ++nt) {
      const int n = nt * 16 + nlane;                 // codeword column for this lane
      const int ntile = (nt + 1) & (NTILES - 1);     // wrapped -> branch-free, in-bounds
      const float e2nxt = e2p[ntile * 16];           // pipelined one tile ahead
      const size_t nxt = (size_t)(ntile * 16) * D_DIM;
      v8f c0 = {}, c1 = {}, c2 = {};                 // inline-0 C seeds all 3 chains
#pragma unroll
      for (int kki = 0; kki < 4; ++kki) {
        v16bf bh = __builtin_bit_cast(v16bf, sh[kki]);
        v16bf bl = __builtin_bit_cast(v16bf, sl[kki]);
        c0 = wmma_bf16(ahi[kki], bh, c0);  // hi*hi
        c1 = wmma_bf16(ahi[kki], bl, c1);  // hi*lo
        c2 = wmma_bf16(alo[kki], bh, c2);  // lo*hi
        // reload this slot with next tile's fragment (consumed above -> pipelined)
        sh[kki] = *(const v16us*)(ph + nxt + kki * 32);
        sl[kki] = *(const v16us*)(pl + nxt + kki * 32);
      }
#pragma unroll
      for (int i = 0; i < 8; ++i) {
        float s = (c0[i] + c1[i]) + (c2[i] + e2cur);  // = 0.5||e||^2 - r.e, e2 added late
        if (s < vmin[i]) { vmin[i] = s; vidx[i] = n; }
      }
      e2cur = e2nxt;
    }

    // argmin across the 16 lanes of each half (C layout: lanes0-15 -> M=v, lanes16-31 -> M=v+8)
#pragma unroll
    for (int mask = 8; mask >= 1; mask >>= 1) {
#pragma unroll
      for (int i = 0; i < 8; ++i) {
        float om = __shfl_xor(vmin[i], mask, 32);
        int   oi = __shfl_xor(vidx[i], mask, 32);
        if (om < vmin[i] || (om == vmin[i] && oi < vidx[i])) { vmin[i] = om; vidx[i] = oi; }
      }
    }

    // epilogue: write codes, subtract exact f32 codeword from wave-private rows
#pragma unroll
    for (int r = 0; r < 16; ++r) {
      const int idx = __shfl(vidx[r & 7], (r < 8) ? 0 : 16, 32);
      if (lane == 0) codes[(row0 + m_base + r) * L_LEVELS + l] = idx;
      const float4 qv = ((const float4*)(cb + ((size_t)l * K_CODES + idx) * D_DIM))[lane];
      float4* dst = (float4*)&s_res[(m_base + r) * PITCH];
      float4 cur = dst[lane];
      cur.x -= qv.x; cur.y -= qv.y; cur.z -= qv.z; cur.w -= qv.w;
      dst[lane] = cur;
    }
    __syncthreads();
  }

  // quantized = res_in - res_final (telescoping sum of gathered codewords)
#pragma unroll
  for (int i = 0; i < (ROWS_PER_BLOCK * D_DIM / 4) / THREADS; ++i) {
    int linear = i * THREADS + tid;
    int r = linear >> 5;
    int c4 = linear & 31;
    const float4 a = ((const float4*)(res_in + (row0 + r) * D_DIM))[c4];
    const float4 b = *(const float4*)&s_res[r * PITCH + c4 * 4];
    float4 o;
    o.x = a.x - b.x; o.y = a.y - b.y; o.z = a.z - b.z; o.w = a.w - b.w;
    ((float4*)(qout + (row0 + r) * D_DIM))[c4] = o;
  }
}

extern "C" void kernel_launch(void* const* d_in, const int* in_sizes, int n_in,
                              void* d_out, int out_size, void* d_ws, size_t ws_size,
                              hipStream_t stream) {
  (void)n_in; (void)out_size; (void)ws_size;
  const float* res = (const float*)d_in[0];   // [B, 128] f32
  const float* cb  = (const float*)d_in[1];   // [4, 1024, 128] f32
  const int B = in_sizes[0] / D_DIM;          // 131072
  const int rowsLK = in_sizes[1] / D_DIM;     // 4096 = L*K

  unsigned short* bhi = (unsigned short*)d_ws;                    // 1 MB
  unsigned short* blo = bhi + (size_t)rowsLK * D_DIM;             // 1 MB
  float* e2 = (float*)(blo + (size_t)rowsLK * D_DIM);             // 16 KB

  float* qout = (float*)d_out;                                    // [B,128] f32
  int* codes = (int*)((float*)d_out + (size_t)B * D_DIM);         // [B,4] i32

  rvq_prep_67010079752343<<<rowsLK, 128, 0, stream>>>(cb, bhi, blo, e2);
  rvq_main_67010079752343<<<B / ROWS_PER_BLOCK, THREADS, 0, stream>>>(
      res, cb, bhi, blo, e2, qout, codes);
}
